// OfflineSlidingWindowAttn_32701880992151
// MI455X (gfx1250) — matrix operand order repaired and checked by reading, not compiled, not run on hardware
//
#include <hip/hip_runtime.h>

// CDNA5 / gfx1250 sliding-window GQA attention, fp32 io, f16 WMMA compute.
// One wave32 per 16-query tile; two-phase softmax because the reference
// applies clip() AFTER normalization. Logit strip cached in LDS (fp32).

typedef __attribute__((ext_vector_type(16))) _Float16 v16h;
typedef __attribute__((ext_vector_type(8)))  float    v8f;

#define HQ    32
#define HKV   8
#define DHEAD 128
#define WIN   512

__device__ __forceinline__ unsigned pkh(float a, float b) {
  union { _Float16 h[2]; unsigned u; } x;
  x.h[0] = (_Float16)a; x.h[1] = (_Float16)b;
  return x.u;
}

// branch-free tanh: 1 - 2/(exp(2x)+1); exp->inf gives +1, exp->0 gives -1.
__device__ __forceinline__ float tanh_from_2x(float x2) {
  float e = __expf(x2);                       // v_exp_f32 (TRANS)
  return 1.0f - 2.0f * __builtin_amdgcn_rcpf(e + 1.0f);  // v_rcp_f32 (TRANS)
}

union HV {
  v16h v;
  unsigned u[8];
  uint4 q4[2];
};

__global__ __launch_bounds__(32)
void swa_fwd(const float* __restrict__ Q, const float* __restrict__ K,
             const float* __restrict__ V, float* __restrict__ O,
             int B, int SQ, int SK)
{
  constexpr int SCOLS = 576;            // max window span (512+16+align) is 544; padded
  constexpr int VTS   = 20;             // dword stride per d-row of V^T tile
  __shared__ float    sS[16 * SCOLS];   // 36 KB capped+masked logits
  __shared__ unsigned sVT[DHEAD * VTS]; // 10 KB V^T f16 tile (32 keys)
  __shared__ float    sM[16], sL[16];

  const int lane = threadIdx.x & 31;
  const int half = lane >> 4;   // which 16-lane half
  const int l16  = lane & 15;

  int wg = blockIdx.x;
  const int qtiles = SQ >> 4;
  const int q0 = (wg % qtiles) << 4;  wg /= qtiles;
  const int h  = wg % HQ;             wg /= HQ;
  const int b  = wg;
  const int hkv = h / (HQ / HKV);

  const float scale = 0.088388347648318447f;     // 1/sqrt(128)
  const float cap   = 30.0f;
  const float c2x   = 2.0f * scale / 30.0f;      // raw_s -> 2*(s*scale/cap)

  const long long qrow = (long long)HQ  * DHEAD;
  const long long krow = (long long)HKV * DHEAD;

  // ---------------- Q tile -> WMMA A layout (f16), kept in regs ------------
  // A layout: lane row m=l16; elements 0..7 -> k=8*half+e, 8..15 -> k=16+8*half+e
  HV qa[4];
  {
    const float* qp = Q + ((long long)b * SQ + (q0 + l16)) * qrow + (long long)h * DHEAD;
#pragma unroll
    for (int c = 0; c < 4; ++c) {
      const float* p0 = qp + c * 32 + 8 * half;
      const float* p1 = qp + c * 32 + 16 + 8 * half;
      float4 f0 = *(const float4*)(p0);
      float4 f1 = *(const float4*)(p0 + 4);
      float4 f2 = *(const float4*)(p1);
      float4 f3 = *(const float4*)(p1 + 4);
      qa[c].u[0] = pkh(f0.x, f0.y); qa[c].u[1] = pkh(f0.z, f0.w);
      qa[c].u[2] = pkh(f1.x, f1.y); qa[c].u[3] = pkh(f1.z, f1.w);
      qa[c].u[4] = pkh(f2.x, f2.y); qa[c].u[5] = pkh(f2.z, f2.w);
      qa[c].u[6] = pkh(f3.x, f3.y); qa[c].u[7] = pkh(f3.z, f3.w);
    }
  }

  int lo = q0 - WIN; if (lo < 0) lo = 0;
  const int kv0   = lo & ~31;
  const int kvend = q0 + 16;
  const int n32   = (kvend - kv0 + 31) >> 5;
  const int n16   = n32 * 2;

  float Mr_[8], Ls_[8];
#pragma unroll
  for (int r = 0; r < 8; ++r) { Mr_[r] = -1e30f; Ls_[r] = 0.0f; }

  // ====== phase 1: S = mask(cap*tanh(scale*(Q K^T)/cap)); cache + stats ====
  for (int t = 0; t < n16; ++t) {
    const int j0 = kv0 + t * 16;
    int jc = j0 + l16; if (jc > SK - 1) jc = SK - 1;
    const float* kp = K + ((long long)b * SK + jc) * krow + (long long)hkv * DHEAD;

    v8f s = {};
#pragma unroll
    for (int c = 0; c < 4; ++c) {
      // B layout: lane col n=l16 (key), elements 0..7 -> k=8*half+e, 8..15 -> k=16+8*half+e
      const float* p = kp + c * 32 + 8 * half;
      float4 f0 = *(const float4*)(p);
      float4 f1 = *(const float4*)(p + 4);
      float4 f2 = *(const float4*)(p + 16);
      float4 f3 = *(const float4*)(p + 20);
      HV kb;
      kb.u[0] = pkh(f0.x, f0.y); kb.u[1] = pkh(f0.z, f0.w);
      kb.u[2] = pkh(f1.x, f1.y); kb.u[3] = pkh(f1.z, f1.w);
      kb.u[4] = pkh(f2.x, f2.y); kb.u[5] = pkh(f2.z, f2.w);
      kb.u[6] = pkh(f3.x, f3.y); kb.u[7] = pkh(f3.z, f3.w);
      s = __builtin_amdgcn_wmma_f32_16x16x32_f16(false, qa[c].v, false, kb.v,
                                                 (short)0, s, false, false);
    }

    const int j = j0 + l16;
    const int colbase = j0 - kv0;
#pragma unroll
    for (int r = 0; r < 8; ++r) {
      const int m = r + 8 * half;     // C layout: vgpr r, half -> row m
      const int i = q0 + m;
      float sv = cap * tanh_from_2x(s[r] * c2x);
      // sliding-window causal mask: 0 <= i-j <= WIN
      const bool ok = ((unsigned)(i - j) <= (unsigned)WIN);
      sv = ok ? sv : -1e30f;
      sS[m * SCOLS + colbase + l16] = sv;
      s[r] = sv;
    }

    // online row max / sum-of-exp (reduce over the 16 N-lanes of each half)
#pragma unroll
    for (int r = 0; r < 8; ++r) {
      float rmax = s[r];
#pragma unroll
      for (int d = 1; d < 16; d <<= 1) rmax = fmaxf(rmax, __shfl_xor(rmax, d, 32));
      const float Mn = fmaxf(Mr_[r], rmax);
      float p = (s[r] <= -1e29f) ? 0.0f : __expf(s[r] - Mn);
#pragma unroll
      for (int d = 1; d < 16; d <<= 1) p += __shfl_xor(p, d, 32);
      const float corr = (Mr_[r] <= -1e29f) ? 0.0f : __expf(Mr_[r] - Mn);
      Ls_[r] = Ls_[r] * corr + p;
      Mr_[r] = Mn;
    }
  }

  if (l16 == 0) {
#pragma unroll
    for (int r = 0; r < 8; ++r) { sM[r + 8 * half] = Mr_[r]; sL[r + 8 * half] = Ls_[r]; }
  }
  asm volatile("s_wait_dscnt 0x0" ::: "memory");

  const float Mrow = sM[l16];
  const float rL   = __builtin_amdgcn_rcpf(fmaxf(sL[l16], 1e-30f));

  // ====== phase 2: O = clip(exp(S-M)/L) * V (32 keys per step) =============
  v8f zz = {};
  v8f acc[8];
#pragma unroll
  for (int c = 0; c < 8; ++c) acc[c] = zz;

  for (int t = 0; t < n32; ++t) {
    const int j0 = kv0 + t * 32;

    // stage V^T (f16) tile: sVT[d*VTS + keypair], lanes sweep d (coalesced)
#pragma unroll 4
    for (int kp2 = 0; kp2 < 16; ++kp2) {
      int ja = j0 + 2 * kp2, jb = ja + 1;
      if (ja > SK - 1) ja = SK - 1;
      if (jb > SK - 1) jb = SK - 1;
      const float* va = V + ((long long)b * SK + ja) * krow + (long long)hkv * DHEAD;
      const float* vb = V + ((long long)b * SK + jb) * krow + (long long)hkv * DHEAD;
#pragma unroll
      for (int rep = 0; rep < 4; ++rep) {
        const int d = lane + rep * 32;
        sVT[d * VTS + kp2] = pkh(va[d], vb[d]);
      }
    }
    // RAW: staged V^T must land before B-tile reads (single wave, no barrier)
    asm volatile("s_wait_dscnt 0x0" ::: "memory");

    // P tile in A layout straight from the LDS S-cache
    HV pa;
    {
      const float* srow = &sS[l16 * SCOLS + (j0 - kv0)];
      float4 r0 = *(const float4*)(srow + 8 * half);
      float4 r1 = *(const float4*)(srow + 8 * half + 4);
      float4 r2 = *(const float4*)(srow + 16 + 8 * half);
      float4 r3 = *(const float4*)(srow + 16 + 8 * half + 4);
      float e[16] = { r0.x,r0.y,r0.z,r0.w, r1.x,r1.y,r1.z,r1.w,
                      r2.x,r2.y,r2.z,r2.w, r3.x,r3.y,r3.z,r3.w };
#pragma unroll
      for (int ii = 0; ii < 8; ++ii) {
        float p0 = __expf(e[2*ii]   - Mrow) * rL;
        float p1 = __expf(e[2*ii+1] - Mrow) * rL;
        p0 = fminf(fmaxf(1.02f * p0 - 0.01f, 0.0f), 1.0f);
        p1 = fminf(fmaxf(1.02f * p1 - 0.01f, 0.0f), 1.0f);
        pa.u[ii] = pkh(p0, p1);
      }
    }

#pragma unroll
    for (int c = 0; c < 8; ++c) {
      HV vbm;  // B layout: col n=l16 -> d=16c+l16; elements 0..7 -> key 8*half+e, 8..15 -> key 16+8*half+e
      const unsigned* vt = &sVT[(16 * c + l16) * VTS + 4 * half];
      vbm.q4[0] = *(const uint4*)(vt);
      vbm.q4[1] = *(const uint4*)(vt + 8);
      acc[c] = __builtin_amdgcn_wmma_f32_16x16x32_f16(false, pa.v, false, vbm.v,
                                                      (short)0, acc[c], false, false);
    }
    // no trailing wait: per-wave LDS ops are processed in order, so next
    // iteration's sVT stores cannot pass this iteration's reads (WAR safe).
  }

  // ---------------- write O, fp32, layout [b, sq, hq, d] -------------------
  float* op = O + ((long long)b * SQ + q0) * qrow + (long long)h * DHEAD + l16;
#pragma unroll
  for (int c = 0; c < 8; ++c) {
#pragma unroll
    for (int r = 0; r < 8; ++r) {
      const int m = r + 8 * half;
      op[(long long)m * qrow + 16 * c] = acc[c][r];
    }
  }
}

extern "C" void kernel_launch(void* const* d_in, const int* in_sizes, int n_in,
                              void* d_out, int out_size, void* d_ws, size_t ws_size,
                              hipStream_t stream) {
  const float* q = (const float*)d_in[0];
  const float* k = (const float*)d_in[1];
  const float* v = (const float*)d_in[2];
  float* o = (float*)d_out;
  const int SQ = 2048, SK = 2048;                 // per reference setup
  int B = in_sizes[0] / (SQ * HQ * DHEAD);
  if (B < 1) B = 1;
  dim3 grid(B * HQ * (SQ / 16));
  swa_fwd<<<grid, dim3(32), 0, stream>>>(q, k, v, o, B, SQ, SK);
}
